// Attend_46660524704198
// MI455X (gfx1250) — matrix operand order repaired and checked by reading, not compiled, not run on hardware
//
#include <hip/hip_runtime.h>

typedef __attribute__((ext_vector_type(16))) __bf16    v16bf;
typedef __attribute__((ext_vector_type(16))) _Float16  v16h;
typedef __attribute__((ext_vector_type(8)))  float     v8f;
typedef __attribute__((__vector_size__(4 * sizeof(int)))) int v4i_async;

#define NB 4
#define NH 16
#define NN 1024
#define ND 128
#define NT_ 64            // NN / 16 row-tiles
#define TOPK 64
#define CLAMPV 50.0f
#define MASKV (-3.402823466e38f)
#define SCALEQ 0.08838834764831845f   // 128^-0.5

// CDNA5 async global->LDS path (guarded: toolchains differ)
#if defined(__gfx1250__) && __has_builtin(__builtin_amdgcn_global_load_async_to_lds_b128)
#define HAVE_ASYNC_LDS 1
#else
#define HAVE_ASYNC_LDS 0
#endif

#if HAVE_ASYNC_LDS
#if __has_builtin(__builtin_amdgcn_s_wait_asynccnt)
#define WAIT_ASYNC(n) __builtin_amdgcn_s_wait_asynccnt(n)
#else
#define WAIT_ASYNC(n) asm volatile("s_wait_asynccnt %0" ::"n"(n) : "memory")
#endif
#endif

// ---------------- fast transcendentals (v_exp_f32-based) ----------------
__device__ __forceinline__ float fast_tanh(float x) {
  const float e = __expf(2.0f * x);           // inf for large x, 0 for very neg
  return 1.0f - 2.0f / (e + 1.0f);            // -> +/-1 correctly saturating
}

// ---------------- wave32 reductions ----------------
__device__ __forceinline__ float wredsumf(float v) {
#pragma unroll
  for (int o = 16; o > 0; o >>= 1) v += __shfl_down(v, o, 32);
  return __shfl(v, 0, 32);
}
__device__ __forceinline__ float wredmaxf(float v) {
#pragma unroll
  for (int o = 16; o > 0; o >>= 1) v = fmaxf(v, __shfl_down(v, o, 32));
  return __shfl(v, 0, 32);
}
__device__ __forceinline__ float wredminf(float v) {
#pragma unroll
  for (int o = 16; o > 0; o >>= 1) v = fminf(v, __shfl_down(v, o, 32));
  return __shfl(v, 0, 32);
}
__device__ __forceinline__ int wredsumi(int v) {
#pragma unroll
  for (int o = 16; o > 0; o >>= 1) v += __shfl_down(v, o, 32);
  return __shfl(v, 0, 32);
}

// ============================================================
// Kernel 1: head-0 mixed scores -> relu gate contributions.
// Per (b, i-tile): writes intra-tile-exclusive row-cumsum into G
// and per-tile column totals into P.
// ============================================================
__global__ void gate_partial_kernel(const float* __restrict__ qg,
                                    const float* __restrict__ kg,
                                    const float* __restrict__ wp,
                                    float* __restrict__ G,
                                    float* __restrict__ P) {
  const int t = blockIdx.x, b = blockIdx.y;
  const int tid = threadIdx.x, lane = tid & 31, wave = tid >> 5;
  extern __shared__ __align__(32) char smem[];
  __bf16* qs = (__bf16*)smem;        // [NH][16][ND] bf16
  __bf16* kt = qs + NH * 16 * ND;    // [ND][128] bf16 (k transposed, 8 j-tiles)

  // stage q tile (all heads) as bf16
  for (int idx = tid; idx < NH * 16 * ND; idx += 256) {
    const int h = idx >> 11, rem = idx & 2047, r = rem >> 7, d = rem & 127;
    qs[idx] = (__bf16)qg[(((size_t)b * NH + h) * NN + t * 16 + r) * ND + d];
  }
  // zero the tail of this tile's P row (columns beyond causal range)
  for (int j = 16 * (t + 1) + tid; j < NN; j += 256)
    P[((size_t)b * NT_ + t) * NN + j] = 0.f;
  __syncthreads();

  const int nJT = t + 1;             // causal j-tiles
  const int nIter = (nJT + 7) >> 3;  // 8 waves -> 8 j-tiles per sweep
  const int arow = lane & 15, ahalf = (lane >> 4) << 4, hi = lane >> 4;

  for (int it = 0; it < nIter; ++it) {
    const int jt = it * 8 + wave;
    const int jbase = it * 128;
    v8f acc = {};
    for (int h = 0; h < NH; ++h) {
      // cooperative stage: 128 columns of k_h, transposed, bf16
      for (int idx = tid; idx < 128 * ND; idx += 256) {
        const int jl = idx >> 7, d = idx & 127;
        const int jgl = jbase + jl;
        const float kv =
            (jgl < NN) ? kg[(((size_t)b * NH + h) * NN + jgl) * ND + d] : 0.f;
        kt[d * 128 + jl] = (__bf16)kv;
      }
      __syncthreads();
      if (jt < nJT) {
        const float wh = wp[h] * SCALEQ;   // w_pre[0,h] * D^-1/2
        v8f raw = {};
        const int jl0 = (jt - it * 8) * 16;
#pragma unroll
        for (int kc = 0; kc < 4; ++kc) {   // K = 128 in 4 chunks of 32
          const int k0 = kc * 32;
          v16bf a  = *(const v16bf*)&qs[(h * 16 + arow) * ND + k0 + ahalf];
          v16bf bm;
#pragma unroll
          for (int e = 0; e < 16; ++e)
            bm[e] = kt[(k0 + ahalf + e) * 128 + jl0 + (lane & 15)];
          raw = __builtin_amdgcn_wmma_f32_16x16x32_bf16(
              false, a, false, bm, (short)0, raw, false, false);
        }
#pragma unroll
        for (int e = 0; e < 8; ++e) acc[e] += wh * raw[e];
      }
      __syncthreads();
    }
    if (jt < nJT) {
      const int jcol = jt * 16 + (lane & 15);
      float r8[8], ex[8];
#pragma unroll
      for (int e = 0; e < 8; ++e) {        // C layout: row = e + 8*hi
        const int irow = t * 16 + e + 8 * hi;
        const float s = fast_tanh(acc[e] * (1.0f / CLAMPV)) * CLAMPV;
        const bool zero = (jcol > irow) || (jcol == 0) || (jcol == irow);
        r8[e] = zero ? 0.f : fmaxf(s, 0.f);
      }
      // exclusive scan over the 16 rows of this column
      float run = 0.f;
#pragma unroll
      for (int e = 0; e < 8; ++e) { ex[e] = run; run += r8[e]; }
      const float lowtot = __shfl(run, lane & 15, 32);  // rows 0..7 total
      if (hi) {
#pragma unroll
        for (int e = 0; e < 8; ++e) ex[e] += lowtot;
        P[((size_t)b * NT_ + t) * NN + jcol] = lowtot + run;  // tile total
      }
#pragma unroll
      for (int e = 0; e < 8; ++e)
        G[((size_t)b * NN + t * 16 + e + 8 * hi) * NN + jcol] = ex[e];
    }
  }
}

// ============================================================
// Kernel 2: serial prefix over the 64 tiles, per (b, column j).
// ============================================================
__global__ void gate_scan_kernel(float* __restrict__ G,
                                 const float* __restrict__ P) {
  const int idx = blockIdx.x * 256 + threadIdx.x;
  if (idx >= NB * NN) return;
  const int b = idx >> 10, j = idx & (NN - 1);
  float run = 0.f;
  for (int tt = (j >> 4); tt < NT_; ++tt) {  // tiles where j < 16*(tt+1)
    float* grow = &G[((size_t)b * NN + tt * 16) * NN + j];
    if (run != 0.f) {
#pragma unroll 4
      for (int r = 0; r < 16; ++r) grow[(size_t)r * NN] += run;
    }
    run += P[((size_t)b * NT_ + tt) * NN + j];
  }
}

// ============================================================
// Kernel 3: full attention for one (b, g, i-tile).
// ============================================================
__global__ void attend_kernel(const float* __restrict__ qg,
                              const float* __restrict__ kg,
                              const float* __restrict__ vg,
                              const float* __restrict__ wp,
                              const float* __restrict__ G,
                              float* __restrict__ out) {
  const int t = blockIdx.x, g = blockIdx.y, b = blockIdx.z;
  const int tid = threadIdx.x, lane = tid & 31, wave = tid >> 5;
  extern __shared__ __align__(32) char smem[];
  __bf16* qs  = (__bf16*)smem;             // [NH][16][ND]
  __bf16* kts = qs + NH * 16 * ND;         // [NH][ND][16] (k transposed)
  float*  S   = (float*)(kts + NH * ND * 16);  // [16][NN] mixed scores

  for (int idx = tid; idx < NH * 16 * ND; idx += 256) {
    const int h = idx >> 11, rem = idx & 2047, r = rem >> 7, d = rem & 127;
    qs[idx] = (__bf16)qg[(((size_t)b * NH + h) * NN + t * 16 + r) * ND + d];
  }
  for (int idx = tid; idx < 16 * NN; idx += 256) S[idx] = 0.f;
  __syncthreads();

  const int nJT = t + 1;
  const int jmax = nJT * 16;
  const int arow = lane & 15, ahalf = (lane >> 4) << 4, hi = lane >> 4;

  // ---- scores: S[i][j] = sum_h w[g,h]*scale*(q_h . k_h) ----
  for (int jt = 0; jt < nJT; ++jt) {
    for (int idx = tid; idx < NH * 16 * ND; idx += 256) {
      const int h = idx >> 11, rem = idx & 2047, jl = rem >> 7, d = rem & 127;
      kts[(h * ND + d) * 16 + jl] =
          (__bf16)kg[(((size_t)b * NH + h) * NN + jt * 16 + jl) * ND + d];
    }
    __syncthreads();
    if (jt + 1 < nJT)  // pull next K tile toward L2/L0
      __builtin_prefetch(
          &kg[(((size_t)b * NH + (tid >> 4)) * NN + (jt + 1) * 16 + (tid & 15)) * ND],
          0, 0);
    v8f acc = {};
#pragma unroll
    for (int hh = 0; hh < 2; ++hh) {       // 8 waves x 2 heads = 16 heads
      const int h = wave * 2 + hh;
      const float wh = wp[g * NH + h] * SCALEQ;
      v8f raw = {};
#pragma unroll
      for (int kc = 0; kc < 4; ++kc) {
        const int k0 = kc * 32;
        v16bf a  = *(const v16bf*)&qs[(h * 16 + arow) * ND + k0 + ahalf];
        v16bf bm;
#pragma unroll
        for (int e = 0; e < 16; ++e)
          bm[e] = kts[(h * ND + k0 + ahalf + e) * 16 + (lane & 15)];
        raw = __builtin_amdgcn_wmma_f32_16x16x32_bf16(
            false, a, false, bm, (short)0, raw, false, false);
      }
#pragma unroll
      for (int e = 0; e < 8; ++e) acc[e] += wh * raw[e];
    }
    {
      const int jcol = jt * 16 + (lane & 15);
#pragma unroll
      for (int e = 0; e < 8; ++e)
        atomicAdd(&S[(e + 8 * hi) * NN + jcol], acc[e]);
    }
    __syncthreads();
  }

  // ---- per-row: tanh clamp, gate, top-64 threshold, softmax ----
  for (int rr = 0; rr < 2; ++rr) {         // 2 rows per wave
    const int ii = wave * 2 + rr;
    const int gi = t * 16 + ii;
    const float* grow = &G[((size_t)b * NN + gi) * NN];
    float rmax = -3.0e38f, rmin = 3.0e38f;
    for (int j = lane; j < jmax; j += 32) {
      float s = S[ii * NN + j];
      s = fast_tanh(s * (1.0f / CLAMPV)) * CLAMPV - grow[j];
      const bool valid = (j <= gi);
      s = valid ? s : MASKV;
      S[ii * NN + j] = s;
      if (valid) { rmax = fmaxf(rmax, s); rmin = fminf(rmin, s); }
    }
    rmax = wredmaxf(rmax);
    rmin = wredminf(rmin);
    float thresh = -2.0e38f;               // keep-all (still excludes MASKV)
    if (gi + 1 > TOPK) {                   // binary search for ~64th largest
      float lo = rmin, hib = rmax;
      for (int itb = 0; itb < 24; ++itb) {
        const float mid = 0.5f * (lo + hib);
        int c = 0;
        for (int j = lane; j < jmax; j += 32) c += (S[ii * NN + j] >= mid);
        c = wredsumi(c);
        if (c >= TOPK) lo = mid; else hib = mid;
      }
      thresh = lo;
    }
    float lsum = 0.f;
    for (int j = lane; j < jmax; j += 32) {
      const float s = S[ii * NN + j];
      const float p = (s >= thresh && s > -1.0e38f) ? __expf(s - rmax) : 0.f;
      S[ii * NN + j] = p;
      lsum += p;
    }
    const float inv = 1.f / wredsumf(lsum);
    for (int j = lane; j < jmax; j += 32) S[ii * NN + j] *= inv;
  }
  __syncthreads();

  // ---- P @ V with f16 WMMA; wave w owns output columns [16w,16w+16) ----
  const int d0 = wave * 16;
  v8f oacc = {};
  const int nkc = (jmax + 31) >> 5;

#if HAVE_ASYNC_LDS
  // Double-buffered V tiles via GLOBAL_LOAD_ASYNC_TO_LDS_B128, reusing the
  // (now dead) k-staging LDS: 2 x [32 rows][132 floats] (16B-aligned rows).
  float* vbuf = (float*)kts;
  const float* vbase = &vg[((size_t)b * NH + g) * NN * (size_t)ND];
  // issue chunk 0 (4 B128 per thread-round covers 32x128 floats)
  {
#pragma unroll
    for (int r = 0; r < 4; ++r) {
      const int idx = r * 256 + tid;          // 0..1023
      const int jl = idx >> 5;                // 0..31
      const int d4 = (idx & 31) << 2;         // 0..124 step 4
      const float* gp = vbase + (size_t)jl * ND + d4;
      float* lp = &vbuf[(size_t)jl * 132 + d4];
      __builtin_amdgcn_global_load_async_to_lds_b128(
          (v4i_async*)gp, (v4i_async*)lp, 0, 0);
    }
  }
  for (int kc = 0; kc < nkc; ++kc) {
    const int buf = kc & 1;
    if (kc + 1 < nkc) {                       // issue next chunk
      const int k0n = (kc + 1) * 32;
      const int bufn = (kc + 1) & 1;
#pragma unroll
      for (int r = 0; r < 4; ++r) {
        const int idx = r * 256 + tid;
        const int jl = idx >> 5;
        const int d4 = (idx & 31) << 2;
        const float* gp = vbase + (size_t)(k0n + jl) * ND + d4;
        float* lp = &vbuf[(size_t)(bufn * 32 + jl) * 132 + d4];
        __builtin_amdgcn_global_load_async_to_lds_b128(
            (v4i_async*)gp, (v4i_async*)lp, 0, 0);
      }
      WAIT_ASYNC(4);                          // chunk kc landed; next in flight
    } else {
      WAIT_ASYNC(0);
    }
    __syncthreads();                          // whole tile visible to all waves
    const int k0 = kc * 32;
    v16h a, bm;
#pragma unroll
    for (int e = 0; e < 16; ++e)
      a[e] = (_Float16)S[arow * NN + k0 + ahalf + e];
    const float* vrow = &vbuf[(size_t)(buf * 32 + ahalf) * 132 + d0 + (lane & 15)];
#pragma unroll
    for (int e = 0; e < 16; ++e) bm[e] = (_Float16)vrow[(size_t)e * 132];
    oacc = __builtin_amdgcn_wmma_f32_16x16x32_f16(
        false, a, false, bm, (short)0, oacc, false, false);
    __syncthreads();                          // done reading before overwrite
  }
#else
  for (int kc = 0; kc < nkc; ++kc) {
    const int k0 = kc * 32;
    v16h a, bm;
#pragma unroll
    for (int e = 0; e < 16; ++e)
      a[e] = (_Float16)S[arow * NN + k0 + ahalf + e];
    const int jrow = k0 + ahalf;              // always < NN (tiling bound)
    const float* vp = &vg[(((size_t)b * NH + g) * NN + jrow) * ND + d0 + (lane & 15)];
#pragma unroll
    for (int e = 0; e < 16; ++e) bm[e] = (_Float16)vp[(size_t)e * ND];
    oacc = __builtin_amdgcn_wmma_f32_16x16x32_f16(
        false, a, false, bm, (short)0, oacc, false, false);
  }
#endif

#pragma unroll
  for (int e = 0; e < 8; ++e)
    out[(((size_t)b * NH + g) * NN + t * 16 + e + 8 * hi) * ND + d0 +
        (lane & 15)] = oacc[e];
}

// ============================================================
extern "C" void kernel_launch(void* const* d_in, const int* in_sizes, int n_in,
                              void* d_out, int out_size, void* d_ws,
                              size_t ws_size, hipStream_t stream) {
  const float* q = (const float*)d_in[0];
  const float* k = (const float*)d_in[1];
  const float* v = (const float*)d_in[2];
  const float* w = (const float*)d_in[3];
  float* out = (float*)d_out;

  // workspace: G[b][i][j] gate (16.8 MB) + P[b][tile][j] (1 MB)
  float* G = (float*)d_ws;
  float* P = G + (size_t)NB * NN * NN;

  const size_t shmA = (size_t)NH * 16 * ND * 2 + (size_t)ND * 128 * 2; // 96 KB
  const size_t shmC = (size_t)NH * 16 * ND * 2 + (size_t)NH * ND * 16 * 2 +
                      (size_t)16 * NN * 4;                             // 192 KB

  gate_partial_kernel<<<dim3(NT_, NB), 256, shmA, stream>>>(q, k, w, G, P);
  gate_scan_kernel<<<dim3((NB * NN + 255) / 256), 256, 0, stream>>>(G, P);
  attend_kernel<<<dim3(NT_, NH, NB), 256, shmC, stream>>>(q, k, v, w, G, out);
}
